// MLPRepairModule_30056181137664
// MI455X (gfx1250) — matrix operand order repaired and checked by reading, not compiled, not run on hardware
//
#include <hip/hip_runtime.h>
#include <hip/hip_bf16.h>

typedef __attribute__((ext_vector_type(16))) __bf16 v16bf;
typedef __attribute__((ext_vector_type(8)))  float  v8f;

#define Hdim 512
#define Bdim 8
#define Tdim 2048
#define Vdim 32000
#define OUTW (Vdim + Tdim)   // 34048
#define A_STRIDE 520         // padded bf16 row stride for A panel
#define P_STRIDE 520         // padded f32 row stride for proj tile
#define MTILE 32             // rows per workgroup (two 16-row WMMA M-tiles)

// ---------------- prep kernel 1: qb[b,k] = b1[k] + sum_h in[b,h]*w1[k,h] ----------------
__global__ __launch_bounds__(256) void qb_kernel(const float* __restrict__ in_emb,
                                                 const float* __restrict__ w1,
                                                 const float* __restrict__ b1,
                                                 float* __restrict__ qb) {
    int idx = blockIdx.x * 256 + threadIdx.x;      // 0..4095
    int b = idx >> 9, k = idx & 511;
    const float* x = in_emb + b * Hdim;
    const float* w = w1 + (size_t)k * 1024;        // w1[k, h], contiguous in h
    float s = b1[k];
    for (int h = 0; h < Hdim; ++h) s += x[h] * w[h];
    qb[idx] = s;
}

// ---------------- prep kernel 2: Bt[n][h] = bf16(w1[n, 512+h])  (w1t as N-major) --------
__global__ __launch_bounds__(256) void bt_kernel(const float* __restrict__ w1,
                                                 __hip_bfloat16* __restrict__ Bt) {
    int idx = blockIdx.x * 256 + threadIdx.x;      // 0..262143
    int n = idx >> 9, h = idx & 511;
    Bt[idx] = __float2bfloat16(w1[(size_t)n * 1024 + 512 + h]);
}

// ---------------- main fused kernel -----------------------------------------------------
// tile < 1000  : decoder rows v0 = tile*32         (A rows = decoder_weight)
// tile >= 1000 : repair  rows for batch b_own      (A rows = target_embeds[b_own])
__global__ __launch_bounds__(256)
void main_kernel(const float* __restrict__ dec_w, const float* __restrict__ tgt,
                 const float* __restrict__ mask,  const float* __restrict__ w2,
                 const float* __restrict__ qb,    const __hip_bfloat16* __restrict__ Bt,
                 float* __restrict__ out) {
    __shared__ __align__(16) __hip_bfloat16 lds_A[MTILE * A_STRIDE];
    __shared__ __align__(16) float lds_P[MTILE * P_STRIDE];
    __shared__ float lds_qb[Bdim * Hdim];
    __shared__ float lds_w2[Hdim];
    __shared__ float lds_part[256];

    const int tid  = threadIdx.x;
    const int tile = blockIdx.x;
    const bool repair = (tile >= Vdim / MTILE);
    int b_own = 0, t0 = 0;
    const float* arow;
    if (!repair) {
        arow = dec_w + (size_t)tile * MTILE * Hdim;
    } else {
        int rt = tile - Vdim / MTILE;              // 0..511
        b_own  = rt >> 6;
        t0     = (rt & 63) * MTILE;
        arow   = tgt + ((size_t)b_own * Tdim + t0) * Hdim;
    }

    // Stage A panel (32 x 512 f32 -> bf16 in LDS), plus qb and w2.
    for (int idx = tid; idx < MTILE * Hdim; idx += 256) {
        int r = idx >> 9, h = idx & 511;
        lds_A[r * A_STRIDE + h] = __float2bfloat16(arow[(size_t)r * Hdim + h]);
    }
    for (int idx = tid; idx < Bdim * Hdim; idx += 256) lds_qb[idx] = qb[idx];
    for (int idx = tid; idx < Hdim; idx += 256)        lds_w2[idx] = w2[idx];
    __syncthreads();

    const int wave   = tid >> 5;
    const int lane   = tid & 31;
    const int lo     = lane & 15;
    const int hi     = lane >> 4;                  // 0 or 1
    const int n_base = wave * 64;

    // two M-halves x four N-subtiles
    v8f a0n0 = {}, a0n1 = {}, a0n2 = {}, a0n3 = {};
    v8f a1n0 = {}, a1n1 = {}, a1n2 = {}, a1n3 = {};

    // A fragment per-lane base: row m = lo (+16 for second half);
    // lane<16 -> K{0..7,16..23}, lane>=16 -> +8
    const __hip_bfloat16* Arow0 = lds_A + lo * A_STRIDE + hi * 8;
    const __hip_bfloat16* Arow1 = lds_A + (16 + lo) * A_STRIDE + hi * 8;
    // B fragment per-lane base: column n, K{0..7,16..23}; lane>=16 -> +8
    const __hip_bfloat16* Bcol  = Bt + (size_t)(n_base + lo) * Hdim + hi * 8;

    for (int k0 = 0; k0 < Hdim; k0 += 32) {
        union { v16bf v; int4 q[2]; } af0, af1, bf0, bf1, bf2, bf3;
        af0.q[0] = *(const int4*)(Arow0 + k0);
        af0.q[1] = *(const int4*)(Arow0 + k0 + 16);
        af1.q[0] = *(const int4*)(Arow1 + k0);
        af1.q[1] = *(const int4*)(Arow1 + k0 + 16);
        const __hip_bfloat16* bp = Bcol + k0;
        bf0.q[0] = *(const int4*)(bp);                 bf0.q[1] = *(const int4*)(bp + 16);
        bf1.q[0] = *(const int4*)(bp + 16 * Hdim);     bf1.q[1] = *(const int4*)(bp + 16 * Hdim + 16);
        bf2.q[0] = *(const int4*)(bp + 32 * Hdim);     bf2.q[1] = *(const int4*)(bp + 32 * Hdim + 16);
        bf3.q[0] = *(const int4*)(bp + 48 * Hdim);     bf3.q[1] = *(const int4*)(bp + 48 * Hdim + 16);

        a0n0 = __builtin_amdgcn_wmma_f32_16x16x32_bf16(false, af0.v, false, bf0.v, (short)0, a0n0, false, false);
        a0n1 = __builtin_amdgcn_wmma_f32_16x16x32_bf16(false, af0.v, false, bf1.v, (short)0, a0n1, false, false);
        a0n2 = __builtin_amdgcn_wmma_f32_16x16x32_bf16(false, af0.v, false, bf2.v, (short)0, a0n2, false, false);
        a0n3 = __builtin_amdgcn_wmma_f32_16x16x32_bf16(false, af0.v, false, bf3.v, (short)0, a0n3, false, false);
        a1n0 = __builtin_amdgcn_wmma_f32_16x16x32_bf16(false, af1.v, false, bf0.v, (short)0, a1n0, false, false);
        a1n1 = __builtin_amdgcn_wmma_f32_16x16x32_bf16(false, af1.v, false, bf1.v, (short)0, a1n1, false, false);
        a1n2 = __builtin_amdgcn_wmma_f32_16x16x32_bf16(false, af1.v, false, bf2.v, (short)0, a1n2, false, false);
        a1n3 = __builtin_amdgcn_wmma_f32_16x16x32_bf16(false, af1.v, false, bf3.v, (short)0, a1n3, false, false);
    }

    // Scatter accumulators to proj tile per the f32 C/D layout:
    // value r of tile (mh, nt) -> row mh*16 + r + hi*8, col n_base + nt*16 + lo
    #pragma unroll
    for (int r = 0; r < 8; ++r) {
        int m0 = r + hi * 8;
        int m1 = 16 + m0;
        lds_P[m0 * P_STRIDE + n_base +  0 + lo] = a0n0[r];
        lds_P[m0 * P_STRIDE + n_base + 16 + lo] = a0n1[r];
        lds_P[m0 * P_STRIDE + n_base + 32 + lo] = a0n2[r];
        lds_P[m0 * P_STRIDE + n_base + 48 + lo] = a0n3[r];
        lds_P[m1 * P_STRIDE + n_base +  0 + lo] = a1n0[r];
        lds_P[m1 * P_STRIDE + n_base + 16 + lo] = a1n1[r];
        lds_P[m1 * P_STRIDE + n_base + 32 + lo] = a1n2[r];
        lds_P[m1 * P_STRIDE + n_base + 48 + lo] = a1n3[r];
    }
    __syncthreads();

    // Fused epilogue: out = sum_k relu(qb + proj) * w2
    if (!repair) {
        // 256 threads = 8 batches x 32 rows; one output each, full K reduction.
        int bb = tid >> 5, vl = tid & 31;
        const float* qrow = lds_qb + bb * Hdim;
        const float* prow = lds_P + vl * P_STRIDE;
        float s = 0.f;
        for (int k = 0; k < Hdim; ++k) {
            float v = qrow[k] + prow[k];
            v = v > 0.f ? v : 0.f;
            s += v * lds_w2[k];
        }
        out[(size_t)bb * OUTW + (size_t)tile * MTILE + vl] = s;
    } else {
        // 32 rows x 8 k-slices, then 8-way partial reduction per row.
        int vl = tid >> 3, sub = tid & 7;
        const float* qrow = lds_qb + b_own * Hdim;
        const float* prow = lds_P + vl * P_STRIDE;
        float s = 0.f;
        for (int k = sub; k < Hdim; k += 8) {
            float v = qrow[k] + prow[k];
            v = v > 0.f ? v : 0.f;
            s += v * lds_w2[k];
        }
        lds_part[tid] = s;
        __syncthreads();
        if (sub == 0) {
            float tot = 0.f;
            #pragma unroll
            for (int j = 0; j < 8; ++j) tot += lds_part[vl * 8 + j];
            float mv = mask[b_own * Tdim + t0 + vl];
            tot = mv * tot - 1000.f * (1.f - mv);
            out[(size_t)b_own * OUTW + Vdim + t0 + vl] = tot;
        }
    }
}

extern "C" void kernel_launch(void* const* d_in, const int* in_sizes, int n_in,
                              void* d_out, int out_size, void* d_ws, size_t ws_size,
                              hipStream_t stream) {
    const float* in_emb = (const float*)d_in[0];   // (8,1,512)
    const float* tgt    = (const float*)d_in[1];   // (8,2048,512)
    const float* maskp  = (const float*)d_in[2];   // (8,2048)
    const float* w1     = (const float*)d_in[3];   // (512,1024)
    const float* b1     = (const float*)d_in[4];   // (512,)
    const float* w2     = (const float*)d_in[5];   // (512,)
    const float* dec_w  = (const float*)d_in[6];   // (32000,512)
    float* out = (float*)d_out;                    // (8, 34048)

    float* qb = (float*)d_ws;                                          // 16 KB
    __hip_bfloat16* Bt = (__hip_bfloat16*)((char*)d_ws + 16384);       // 512 KB

    qb_kernel<<<16,   256, 0, stream>>>(in_emb, w1, b1, qb);
    bt_kernel<<<1024, 256, 0, stream>>>(w1, Bt);

    const int n_tiles = Vdim / MTILE + (Bdim * Tdim) / MTILE;          // 1000 + 512
    main_kernel<<<n_tiles, 256, 0, stream>>>(dec_w, tgt, maskp, w2, qb, Bt, out);
}
